// ContrastiveLoss_43550968382225
// MI455X (gfx1250) — matrix operand order repaired and checked by reading, not compiled, not run on hardware
//
#include <hip/hip_runtime.h>

// CDNA5 (gfx1250) wave32 WMMA types
typedef __attribute__((ext_vector_type(16))) __bf16 v16bf;
typedef __attribute__((ext_vector_type(8)))  float  v8f;

#define NROWS 4096   // N
#define TOT   8192   // 2N
#define DDIM  128    // feature dim
#define NBLK  64     // 8192 / 128 tile blocks per side

// ---------------------------------------------------------------------------
// Kernel 1: fp32 row L2-normalize -> bf16 Z, plus fp32 positive-pair cosine.
// One block (128 threads = 4 waves) per row pair r.
// ---------------------------------------------------------------------------
__global__ __launch_bounds__(128) void nt_normalize(
    const float* __restrict__ zi, const float* __restrict__ zj,
    __bf16* __restrict__ Zn, float* __restrict__ cosv) {
  const int r = blockIdx.x;     // 0..4095
  const int t = threadIdx.x;    // 0..127
  const float a = zi[(size_t)r * DDIM + t];
  const float b = zj[(size_t)r * DDIM + t];
  float aa = a * a, bb = b * b, ab = a * b;
  #pragma unroll
  for (int off = 16; off > 0; off >>= 1) {
    aa += __shfl_xor(aa, off);
    bb += __shfl_xor(bb, off);
    ab += __shfl_xor(ab, off);
  }
  __shared__ float sh[3][4];
  const int w = t >> 5;
  if ((t & 31) == 0) { sh[0][w] = aa; sh[1][w] = bb; sh[2][w] = ab; }
  __syncthreads();
  aa = sh[0][0] + sh[0][1] + sh[0][2] + sh[0][3];
  bb = sh[1][0] + sh[1][1] + sh[1][2] + sh[1][3];
  ab = sh[2][0] + sh[2][1] + sh[2][2] + sh[2][3];
  const float na = sqrtf(aa), nb = sqrtf(bb);
  Zn[(size_t)r * DDIM + t]           = (__bf16)(a / na);
  Zn[(size_t)(r + NROWS) * DDIM + t] = (__bf16)(b / nb);
  if (t == 0) cosv[r] = ab / fmaxf(na * nb, 1e-8f);  // torch CosineSimilarity eps
}

// ---------------------------------------------------------------------------
// Kernel 2: fused Gram + exp + row/column sums, LOWER TRIANGLE ONLY.
// exp(2*G) is symmetric: each off-diagonal 128x128 tile contributes row sums
// to partial[bx][m-strip] and column sums to partial[by][n-strip] (halves
// WMMA and v_exp work). 4 waves / block, 32 rows per wave (16 accumulators)
// so each B fragment feeds 2 WMMAs -> LDS traffic per WMMA halved.
// Tile fill uses gfx1250 GLOBAL_LOAD_ASYNC_TO_LDS_B128 (ASYNCcnt DMA).
// ---------------------------------------------------------------------------
__global__ __launch_bounds__(128) void nt_gram(
    const __bf16* __restrict__ Zn, float* __restrict__ partial) {
  const int bx = blockIdx.x, by = blockIdx.y;
  if (by > bx) return;                       // lower triangle only
  const bool diagblk = (bx == by);

  __shared__ uint4 As[128][16];   // [row][chunk of 8 bf16], chunk XOR-swizzled
  __shared__ uint4 Bs[128][16];
  const int m0 = by * 128, n0 = bx * 128;
  const int tid = threadIdx.x;    // 0..127

  // Async tile fill: 128 rows x 16 chunks per tile; GVS mode
  // (saddr = Zn base, vaddr = byte offset, vdst = LDS byte address).
  {
    const unsigned long long base = (unsigned long long)Zn;
    for (int i = tid; i < 128 * 16; i += 128) {
      const int row = i >> 4, ch = i & 15;
      const int sw = ch ^ (row & 15);
      const unsigned aoff = (unsigned)(((m0 + row) * DDIM + ch * 8) * 2);
      const unsigned boff = (unsigned)(((n0 + row) * DDIM + ch * 8) * 2);
      // flat addr low 32 bits == wave-relative LDS byte offset (ISA 10.2)
      const unsigned lA = (unsigned)(unsigned long long)&As[row][sw];
      const unsigned lB = (unsigned)(unsigned long long)&Bs[row][sw];
      asm volatile("global_load_async_to_lds_b128 %0, %1, %2 offset:0"
                   :: "v"(lA), "v"(aoff), "s"(base) : "memory");
      asm volatile("global_load_async_to_lds_b128 %0, %1, %2 offset:0"
                   :: "v"(lB), "v"(boff), "s"(base) : "memory");
    }
    asm volatile("s_wait_asynccnt 0x0" ::: "memory");
  }
  __syncthreads();

  const int wave = tid >> 5, lane = tid & 31;
  const int half = lane >> 4, lm = lane & 15;
  const int wr0 = wave * 32;                 // wave's 32-row strip in block

  v8f c[2][8] = {};                          // 2 row-subtiles x 8 col-tiles

  #pragma unroll
  for (int ks = 0; ks < 4; ++ks) {           // K = 128 in steps of 32
    // Two A fragments (16x32 bf16, ISA 7.12.2): lane lm = M;
    // elems 0..7 K=kb+half*8+e, elems 8..15 K=kb+16+half*8+e.
    union { uint4 u[2]; v16bf v; } ua[2];
    #pragma unroll
    for (int s = 0; s < 2; ++s) {
      const int ar = wr0 + s * 16 + lm, asw = ar & 15;
      ua[s].u[0] = As[ar][(ks * 4 + half) ^ asw];
      ua[s].u[1] = As[ar][(ks * 4 + half + 2) ^ asw];
    }
    #pragma unroll
    for (int t = 0; t < 8; ++t) {
      // B fragment (32x16 bf16): lane lm = N; same K order as A:
      // elems 0..7 K=kb+half*8+e, elems 8..15 K=kb+16+half*8+e.
      const int br = t * 16 + lm, bsw = br & 15;
      union { uint4 u[2]; v16bf v; } ub;
      ub.u[0] = Bs[br][(ks * 4 + half) ^ bsw];
      ub.u[1] = Bs[br][(ks * 4 + half + 2) ^ bsw];
      c[0][t] = __builtin_amdgcn_wmma_f32_16x16x32_bf16(
          false, ua[0].v, false, ub.v, (short)0, c[0][t], false, false);
      c[1][t] = __builtin_amdgcn_wmma_f32_16x16x32_bf16(
          false, ua[1].v, false, ub.v, (short)0, c[1][t], false, false);
    }
  }

  // Fused epilogue: e = exp(sim/temp) = exp(2*g); diag excluded on diagonal
  // blocks. C layout: VGPR r, lanes 0-15 -> M=r (N=lm); lanes 16-31 -> M=r+8.
  float rs[2][8], cs[8];
  #pragma unroll
  for (int r = 0; r < 8; ++r) { rs[0][r] = 0.0f; rs[1][r] = 0.0f; cs[r] = 0.0f; }
  #pragma unroll
  for (int t = 0; t < 8; ++t) {
    const int gcol = n0 + t * 16 + lm;
    #pragma unroll
    for (int s = 0; s < 2; ++s) {
      #pragma unroll
      for (int r = 0; r < 8; ++r) {
        const int grow = m0 + wr0 + s * 16 + r + 8 * half;
        float e = __expf(2.0f * c[s][t][r]);
        if (diagblk && grow == gcol) e = 0.0f;
        rs[s][r] += e;      // row accumulation (over columns)
        cs[t] += e;         // column accumulation (over this lane's 16 rows)
      }
    }
  }

  // Row sums: segmented reduction across each 16-lane group (16 columns).
  #pragma unroll
  for (int s = 0; s < 2; ++s) {
    #pragma unroll
    for (int r = 0; r < 8; ++r) {
      #pragma unroll
      for (int off = 1; off < 16; off <<= 1) rs[s][r] += __shfl_xor(rs[s][r], off);
    }
  }
  if (lm == 0) {            // lanes 0 and 16 hold rows r and r+8 of each subtile
    #pragma unroll
    for (int s = 0; s < 2; ++s) {
      #pragma unroll
      for (int r = 0; r < 8; ++r) {
        const int grow = m0 + wr0 + s * 16 + r + 8 * half;
        partial[(size_t)bx * TOT + grow] = rs[s][r];
      }
    }
  }

  // Column sums (off-diagonal blocks only): combine the two half-wave row
  // groups, then reduce the 4 wave strips through re-used LDS.
  if (!diagblk) {
    #pragma unroll
    for (int t = 0; t < 8; ++t) cs[t] += __shfl_xor(cs[t], 16);
    __syncthreads();                         // tiles dead: safe to alias As
    float* colsum = (float*)As;              // 4 waves x 128 cols = 2KB
    if (half == 0) {
      #pragma unroll
      for (int t = 0; t < 8; ++t) colsum[wave * 128 + t * 16 + lm] = cs[t];
    }
    __syncthreads();
    {
      float s = 0.0f;
      #pragma unroll
      for (int w = 0; w < 4; ++w) s += colsum[w * 128 + tid];
      partial[(size_t)by * TOT + n0 + tid] = s;   // symmetry: col sum == row sum
    }
  }
}

// ---------------------------------------------------------------------------
// Kernel 3: den[r] = sum of 64 partials; loss_r = log(den) - cos/temp;
// output mean over 8192 rows. Single block, fixed reduction order.
// ---------------------------------------------------------------------------
__global__ __launch_bounds__(256) void nt_finalize(
    const float* __restrict__ partial, const float* __restrict__ cosv,
    float* __restrict__ out) {
  const int tid = threadIdx.x;
  float acc = 0.0f;
  for (int r = tid; r < TOT; r += 256) {
    float den = 0.0f;
    #pragma unroll 4
    for (int b = 0; b < NBLK; ++b) den += partial[(size_t)b * TOT + r];
    acc += logf(den) - 2.0f * cosv[r & (NROWS - 1)];   // 1/temp = 2
  }
  #pragma unroll
  for (int off = 16; off > 0; off >>= 1) acc += __shfl_xor(acc, off);
  __shared__ float sh[8];
  if ((tid & 31) == 0) sh[tid >> 5] = acc;
  __syncthreads();
  if (tid == 0) {
    float s = 0.0f;
    #pragma unroll
    for (int w = 0; w < 8; ++w) s += sh[w];
    out[0] = s / (float)TOT;
  }
}

// ---------------------------------------------------------------------------
extern "C" void kernel_launch(void* const* d_in, const int* in_sizes, int n_in,
                              void* d_out, int out_size, void* d_ws, size_t ws_size,
                              hipStream_t stream) {
  const float* zi = (const float*)d_in[0];
  const float* zj = (const float*)d_in[1];
  float* out = (float*)d_out;

  // Workspace layout: bf16 Z [8192*128] | cos [4096] | partial [64*8192]
  __bf16* Zn    = (__bf16*)d_ws;
  float* cosv    = (float*)((char*)d_ws + (size_t)TOT * DDIM * sizeof(__bf16));
  float* partial = cosv + NROWS;

  nt_normalize<<<NROWS, 128, 0, stream>>>(zi, zj, Zn, cosv);
  dim3 grid(NBLK, NBLK);
  nt_gram<<<grid, 128, 0, stream>>>(Zn, partial);
  nt_finalize<<<1, 256, 0, stream>>>(partial, cosv, out);
}
